// Network_81209241633451
// MI455X (gfx1250) — matrix operand order repaired and checked by reading, not compiled, not run on hardware
//
#include <hip/hip_runtime.h>
#include <hip/hip_bf16.h>

typedef _Float16 f16;
typedef _Float16 v16h __attribute__((ext_vector_type(16)));
typedef _Float16 v8h  __attribute__((ext_vector_type(8)));
typedef float    v8f  __attribute__((ext_vector_type(8)));

#define NNODES 100000
#define NEDGES 3200000
#define GB     2048

// ---------------- elementwise helpers ----------------
__global__ void k_fill(float* p, float v, int n) {
  int i = blockIdx.x * 256 + threadIdx.x;
  if (i < n) p[i] = v;
}

__global__ void k_degree(const int* __restrict__ dst, float* deg) {
  int e = blockIdx.x * 256 + threadIdx.x;
  if (e < NEDGES) atomicAdd(&deg[dst[e]], 1.0f);
}

__global__ void k_rsqrt_inplace(float* p, int n) {
  int i = blockIdx.x * 256 + threadIdx.x;
  if (i < n) p[i] = rsqrtf(p[i]);
}

// per-edge symmetric norm, computed once and reused by all 3 layers
__global__ void k_norm(const int* __restrict__ src, const int* __restrict__ dst,
                       const float* __restrict__ dinv, float* __restrict__ nrm) {
  int e = blockIdx.x * 256 + threadIdx.x;
  if (e < NEDGES) nrm[e] = dinv[src[e]] * dinv[dst[e]];
}

// one wave32 per edge; lane covers features {lane, lane+32}
__global__ __launch_bounds__(256) void k_scatter(const int* __restrict__ src,
                                                 const int* __restrict__ dst,
                                                 const float* __restrict__ hl,
                                                 const float* __restrict__ nrm,
                                                 float* acc) {
  int e    = (blockIdx.x * 256 + threadIdx.x) >> 5;
  int lane = threadIdx.x & 31;
  if (e >= NEDGES) return;
  int s = src[e], d = dst[e];
  float nr = nrm[e];
  const float* hp = hl + s * 64;
  float* ap = acc + d * 64;
  atomicAdd(&ap[lane],      hp[lane]      * nr);
  atomicAdd(&ap[lane + 32], hp[lane + 32] * nr);
}

// global_add_pool: one wave32 per node
__global__ __launch_bounds__(256) void k_pool(const int* __restrict__ batch,
                                              const float* __restrict__ acc, float* g) {
  int node = (blockIdx.x * 256 + threadIdx.x) >> 5;
  int lane = threadIdx.x & 31;
  if (node >= NNODES) return;
  int b = batch[node];
  atomicAdd(&g[b * 64 + lane],      acc[node * 64 + lane]);
  atomicAdd(&g[b * 64 + lane + 32], acc[node * 64 + lane + 32]);
}

__global__ void k_head2(const float* __restrict__ g2, const float* __restrict__ Wl2,
                        const float* __restrict__ bl2, float* out) {
  int gi = blockIdx.x * 256 + threadIdx.x;
  if (gi < GB) {
    float s = bl2[0];
#pragma unroll
    for (int j = 0; j < 32; ++j) s += g2[gi * 32 + j] * Wl2[j];
    out[gi] = s;
  }
}

// ---------------- WMMA GEMM with fused epilogues ----------------
// Computes out[rowBase .. rowBase+RT*16)[0..NC) = act(A) @ W.
//   RELU_IN : apply relu to A during fp32->f16 staging (fuses previous layer's relu)
//   EPI = 0 : plain store (out = result)
//   EPI = 1 : GCN layer: out = result; acc = bias + result*dinv[row]^2 (self-loop + bias init)
//   EPI = 2 : head: out = relu(result + bias)
// Block = NC/16 waves; wave w computes output columns [16w,16w+16) for all RT row tiles,
// reusing the LDS-resident W^T tile RT times.
template<int K, int NC, int RT, int EPI, bool RELU_IN>
__global__ __launch_bounds__(NC * 2) void k_gemm_wmma(const float* __restrict__ A,
                                                      const float* __restrict__ W,
                                                      float* __restrict__ out,
                                                      const float* __restrict__ dinv,
                                                      const float* __restrict__ bias,
                                                      float* __restrict__ acc) {
  constexpr int NTHR = NC * 2;               // (NC/16 waves) * 32 lanes
  constexpr int ROWS = RT * 16;
  __shared__ f16 lA[ROWS][K];
  __shared__ f16 lWT[NC][K];
  const int tid     = threadIdx.x;
  const int rowBase = blockIdx.x * ROWS;

  // stage A rows (fp32 -> f16, optional fused relu), row-major
  for (int idx = tid; idx < ROWS * K; idx += NTHR) {
    int r = idx / K, k = idx % K;
    float v = A[(rowBase + r) * K + k];
    if (RELU_IN) v = fmaxf(v, 0.0f);
    lA[r][k] = (f16)v;
  }
  // stage W transposed (so B fragments are contiguous in K)
  for (int idx = tid; idx < K * NC; idx += NTHR) {
    int k = idx / NC, c = idx % NC;
    lWT[c][k] = (f16)W[idx];
  }
  __syncthreads();   // all A rows in LDS before any epilogue store may alias A==acc

  const int wave = tid >> 5, lane = tid & 31;
  const int hh = lane >> 4;                  // half-wave index
  const int mn = lane & 15;                  // row (A) / col (B) index
  const int col = wave * 16 + mn;
  const f16* brow = &lWT[col][0];

  for (int t = 0; t < RT; ++t) {
    const int row0 = rowBase + t * 16;
    const f16* arow = &lA[t * 16 + mn][0];
    v8f c = {};
#pragma unroll
    for (int kb = 0; kb < K; kb += 32) {
      // A 16x32 f16 layout: lane half hh holds K = kb+8*hh+{0..7} and kb+16+8*hh+{0..7}
      v8h alo = *(const v8h*)(arow + kb + 8 * hh);
      v8h ahi = *(const v8h*)(arow + kb + 16 + 8 * hh);
      v16h a = __builtin_shufflevector(alo, ahi, 0,1,2,3,4,5,6,7,8,9,10,11,12,13,14,15);
      // B 32x16 f16 layout: lane half hh holds K = kb+8*hh+{0..7} and kb+16+8*hh+{0..7} for column mn
      v8h blo = *(const v8h*)(brow + kb + 8 * hh);
      v8h bhi = *(const v8h*)(brow + kb + 16 + 8 * hh);
      v16h b = __builtin_shufflevector(blo, bhi, 0,1,2,3,4,5,6,7,8,9,10,11,12,13,14,15);
      // (neg_a, A, neg_b, B, c_mod, C, reuse_a, reuse_b)
      c = __builtin_amdgcn_wmma_f32_16x16x32_f16(false, a, false, b, (short)0, c, false, false);
    }
    // C/D layout: VGPR r of lane = element (M = r + 8*hh, N = mn)
#pragma unroll
    for (int r = 0; r < 8; ++r) {
      int m = r + 8 * hh;
      int o = (row0 + m) * NC + col;
      if (EPI == 0) {
        out[o] = c[r];
      } else if (EPI == 1) {
        float di = dinv[row0 + m];
        out[o] = c[r];
        acc[o] = bias[col] + c[r] * di * di;
      } else {
        out[o] = fmaxf(c[r] + bias[col], 0.0f);
      }
    }
  }
}

// ---------------- driver ----------------
extern "C" void kernel_launch(void* const* d_in, const int* in_sizes, int n_in,
                              void* d_out, int out_size, void* d_ws, size_t ws_size,
                              hipStream_t stream) {
  const float* x    = (const float*)d_in[0];
  const int*   ei   = (const int*)d_in[1];    // [2, E]: row 0 = src, row 1 = dst
  const int*   batch= (const int*)d_in[2];
  const float* W1 = (const float*)d_in[3];  const float* b1 = (const float*)d_in[4];
  const float* W2 = (const float*)d_in[5];  const float* b2 = (const float*)d_in[6];
  const float* W3 = (const float*)d_in[7];  const float* b3 = (const float*)d_in[8];
  const float* Wl1= (const float*)d_in[9];  const float* bl1= (const float*)d_in[10];
  const float* Wl2= (const float*)d_in[11]; const float* bl2= (const float*)d_in[12];
  float* out = (float*)d_out;

  // workspace layout (floats)
  float* ws   = (float*)d_ws;
  float* dinv = ws;                          // N
  float* nrm  = dinv + NNODES;               // E    (per-edge norm)
  float* hl   = nrm + NEDGES;                // N*64 (linear-transform output)
  float* acc  = hl + NNODES * 64;            // N*64 (aggregation accumulator)
  float* gp   = acc + NNODES * 64;           // G*64 (pooled graph features)
  float* g2   = gp + GB * 64;                // G*32

  const int* src = ei;
  const int* dst = ei + NEDGES;

  const int eb = (NEDGES + 255) / 256;
  const int nb = (NNODES + 255) / 256;
  const int sb = (NEDGES * 32 + 255) / 256;  // wave-per-edge scatter
  const int pb = (NNODES * 32 + 255) / 256;  // wave-per-node pool
  const int gemm_blocks = NNODES / 80;       // RT=5 -> 80 rows/block -> 1250 blocks

  // degree (init 1.0 for self-loop) -> dinv = rsqrt(deg) -> per-edge norm
  k_fill<<<nb, 256, 0, stream>>>(dinv, 1.0f, NNODES);
  k_degree<<<eb, 256, 0, stream>>>(dst, dinv);
  k_rsqrt_inplace<<<nb, 256, 0, stream>>>(dinv, NNODES);
  k_norm<<<eb, 256, 0, stream>>>(src, dst, dinv, nrm);

  // ---- layer 1: hl = x@W1; acc = b1 + hl*dinv^2; scatter ----
  k_gemm_wmma<128, 64, 5, 1, false><<<gemm_blocks, 128, 0, stream>>>(x, W1, hl, dinv, b1, acc);
  k_scatter<<<sb, 256, 0, stream>>>(src, dst, hl, nrm, acc);

  // ---- layer 2: hl = relu(acc)@W2 (relu fused into staging); acc = b2 + hl*dinv^2; scatter ----
  k_gemm_wmma<64, 64, 5, 1, true><<<gemm_blocks, 128, 0, stream>>>(acc, W2, hl, dinv, b2, acc);
  k_scatter<<<sb, 256, 0, stream>>>(src, dst, hl, nrm, acc);

  // ---- layer 3 (no output relu) ----
  k_gemm_wmma<64, 64, 5, 1, true><<<gemm_blocks, 128, 0, stream>>>(acc, W3, hl, dinv, b3, acc);
  k_scatter<<<sb, 256, 0, stream>>>(src, dst, hl, nrm, acc);

  // ---- global_add_pool ----
  k_fill<<<(GB * 64 + 255) / 256, 256, 0, stream>>>(gp, 0.0f, GB * 64);
  k_pool<<<pb, 256, 0, stream>>>(batch, acc, gp);

  // ---- head: g2 = relu(gp@Wl1 + bl1) (fused epilogue); out = g2@Wl2 + bl2 ----
  k_gemm_wmma<64, 32, 4, 2, false><<<GB / 64, 64, 0, stream>>>(gp, Wl1, g2, dinv, bl1, nullptr);
  k_head2<<<(GB + 255) / 256, 256, 0, stream>>>(g2, Wl2, bl2, out);
}
